// SlidingWindowAttention_11330123727022
// MI455X (gfx1250) — matrix operand order repaired and checked by reading, not compiled, not run on hardware
//
#include <hip/hip_runtime.h>

typedef __attribute__((ext_vector_type(16))) _Float16 v16h;
typedef __attribute__((ext_vector_type(8)))  float    v8f;
typedef __attribute__((ext_vector_type(4)))  float    f32x4;

#define NB    2
#define NH    16
#define NS    2048
#define NDK   128
#define NDV   128
#define NWIN  128
#define NSINK 4
#define MAXC  176   // max band columns (10 tiles * 16 = 160, padded alloc)

__device__ __forceinline__ v8f wmma_f16(v16h a, v16h b, v8f c) {
  // D = A*B + C, 16x16 f32 accum, K=32 f16
  return __builtin_amdgcn_wmma_f32_16x16x32_f16(false, a, false, b, (short)0, c,
                                                false, false);
}

__global__ __launch_bounds__(32) void swa_sink_attn_kernel(
    const float* __restrict__ Q, const float* __restrict__ K,
    const float* __restrict__ V, float* __restrict__ O, float* __restrict__ W) {
  __shared__ float sP[16][MAXC];  // score band -> prob band
  __shared__ int   sJ[MAXC];      // band column -> global key index

  const int lane = threadIdx.x;   // 0..31 (one full wave; EXEC all ones)
  const int g    = lane >> 4;     // half-wave group
  const int lr   = lane & 15;
  const int bh   = blockIdx.z * NH + blockIdx.y;
  const int q0   = blockIdx.x * 16;

  const float* Qb = Q + ((size_t)bh * NS + q0) * NDK;
  const float* Kb = K + (size_t)bh * NS * NDK;
  const float* Vb = V + (size_t)bh * NS * NDV;
  float* Ob = O + ((size_t)bh * NS + q0) * NDV;
  float* Wb = W + ((size_t)bh * NS + q0) * NS;

  const float scale = 0.088388347648318447f;  // 1/sqrt(128)

  // ---- Q A-fragments: 4 K-chunks of 32, f32 -> f16 ----------------------
  // 16-bit A layout: lane row M = lr; element e -> k = e + (e>=8 ? 8:0) + g*8
  v16h aQ[4];
  {
    const float* qrow = Qb + lr * NDK;
#pragma unroll
    for (int kc = 0; kc < 4; ++kc) {
      const int kb = kc * 32 + g * 8;
      f32x4 lo0 = *(const f32x4*)(qrow + kb);
      f32x4 lo1 = *(const f32x4*)(qrow + kb + 4);
      f32x4 hi0 = *(const f32x4*)(qrow + kb + 16);
      f32x4 hi1 = *(const f32x4*)(qrow + kb + 20);
#pragma unroll
      for (int e = 0; e < 4; ++e) {
        aQ[kc][e]      = (_Float16)lo0[e];
        aQ[kc][e + 4]  = (_Float16)lo1[e];
        aQ[kc][e + 8]  = (_Float16)hi0[e];
        aQ[kc][e + 12] = (_Float16)hi1[e];
      }
    }
  }

  // ---- key-tile enumeration: optional sink tile + window tiles ----------
  int tlo = q0 - NWIN + 1; if (tlo < 0) tlo = 0; tlo >>= 4;
  const int hasSink    = (tlo > 0) ? 1 : 0;
  const int numTiles   = hasSink + ((int)blockIdx.x - tlo + 1);  // <= 10
  const int numCols    = numTiles * 16;
  const int numColsPad = (numCols + 31) & ~31;

  // ---- QK^T per key tile, mask, stash band in LDS -----------------------
  for (int t = 0; t < numTiles; ++t) {
    const int jbase = (hasSink && t == 0) ? 0 : (tlo + t - hasSink) * 16;
    const int j     = jbase + lr;  // this lane's key row (B column N = lr)
    // B layout: lane col N = lr; element e -> d = kc*32 + g*8 + e + (e>=8 ? 8:0)
    const float* krow = Kb + (size_t)j * NDK + g * 8;
    v8f c = {};
#pragma unroll
    for (int kc = 0; kc < 4; ++kc) {
      f32x4 k0 = *(const f32x4*)(krow + kc * 32);
      f32x4 k1 = *(const f32x4*)(krow + kc * 32 + 4);
      f32x4 k2 = *(const f32x4*)(krow + kc * 32 + 16);
      f32x4 k3 = *(const f32x4*)(krow + kc * 32 + 20);
      v16h bK;
#pragma unroll
      for (int e = 0; e < 4; ++e) {
        bK[e]      = (_Float16)k0[e];
        bK[e + 4]  = (_Float16)k1[e];
        bK[e + 8]  = (_Float16)k2[e];
        bK[e + 12] = (_Float16)k3[e];
      }
      c = wmma_f16(aQ[kc], bK, c);
    }
    // C layout: VGPR r -> row r + 8g, lane col = lr
#pragma unroll
    for (int r = 0; r < 8; ++r) {
      const int  i  = q0 + r + 8 * g;
      const bool ok = (j <= i) && ((j < NSINK) || (j > i - NWIN));
      sP[r + 8 * g][t * 16 + lr] = ok ? (c[r] * scale) : -1.0e9f;
    }
    sJ[t * 16 + lr] = j;  // duplicate writes (g=0/1) carry identical values
  }
  if (numColsPad > numCols) {  // zero-pad band to a multiple of 32 keys
#pragma unroll
    for (int r = 0; r < 8; ++r) sP[r + 8 * g][numCols + lr] = 0.0f;
    sJ[numCols + lr] = 0;
  }
  __syncthreads();

  // ---- exact row softmax (each half-wave redundantly owns row lr) -------
  float m = -3.0e38f;
  for (int ci = 0; ci < numCols; ++ci) m = fmaxf(m, sP[lr][ci]);
  float sum = 0.0f;
  for (int ci = 0; ci < numCols; ++ci) sum += __expf(sP[lr][ci] - m);
  const float inv = 1.0f / sum;
  for (int ci = 0; ci < numCols; ++ci) {
    const float p = __expf(sP[lr][ci] - m) * inv;  // masked -1e9 -> exactly 0
    sP[lr][ci] = p;
  }
  __syncthreads();

  // ---- write attention-weight band (rest of row pre-zeroed by memset) ---
  // Non-temporal: 512 MB one-shot stream; keep K/V resident in the 192 MB L2.
  for (int t = 0; t < numTiles; ++t) {
    const int jb = (hasSink && t == 0) ? 0 : (tlo + t - hasSink) * 16;
    f32x4 w0 = *(const f32x4*)&sP[lr][t * 16 + g * 8];
    f32x4 w1 = *(const f32x4*)&sP[lr][t * 16 + g * 8 + 4];
    __builtin_nontemporal_store(w0, (f32x4*)(Wb + (size_t)lr * NS + jb + g * 8));
    __builtin_nontemporal_store(w1, (f32x4*)(Wb + (size_t)lr * NS + jb + g * 8 + 4));
  }

  // ---- O = P @ V --------------------------------------------------------
  const int numA = numColsPad >> 5;  // <= 5 K-chunks of 32 band keys
  v8f oacc[8];
#pragma unroll
  for (int nt = 0; nt < 8; ++nt) { v8f z = {}; oacc[nt] = z; }

  for (int ac = 0; ac < numA; ++ac) {
    // A-fragment of P from LDS
    v16h aP;
    const int kb = ac * 32 + g * 8;
#pragma unroll
    for (int e = 0; e < 8; ++e)  aP[e] = (_Float16)sP[lr][kb + e];
#pragma unroll
    for (int e = 8; e < 16; ++e) aP[e] = (_Float16)sP[lr][kb + 8 + e];
    // V row pointers for this K-chunk: element e -> key sJ[ac*32 + g*8 + e + (e>=8 ? 8:0)]
    const float* vp[16];
#pragma unroll
    for (int e = 0; e < 16; ++e)
      vp[e] = Vb + (size_t)sJ[ac * 32 + g * 8 + e + (e >= 8 ? 8 : 0)] * NDV + lr;
#pragma unroll
    for (int nt = 0; nt < 8; ++nt) {
      v16h bV;
#pragma unroll
      for (int e = 0; e < 16; ++e) bV[e] = (_Float16)vp[e][nt * 16];
      oacc[nt] = wmma_f16(aP, bV, oacc[nt]);
    }
  }

  // ---- store O (C layout: VGPR r -> row r+8g, lane -> col), NT stream ---
#pragma unroll
  for (int nt = 0; nt < 8; ++nt)
#pragma unroll
    for (int r = 0; r < 8; ++r)
      __builtin_nontemporal_store(oacc[nt][r],
                                  Ob + (size_t)(r + 8 * g) * NDV + nt * 16 + lr);
}

extern "C" void kernel_launch(void* const* d_in, const int* in_sizes, int n_in,
                              void* d_out, int out_size, void* d_ws, size_t ws_size,
                              hipStream_t stream) {
  (void)in_sizes; (void)n_in; (void)out_size; (void)d_ws; (void)ws_size;
  const float* Q = (const float*)d_in[0];
  const float* K = (const float*)d_in[1];
  const float* V = (const float*)d_in[2];
  float* out = (float*)d_out;
  float* O = out;                                      // [B,H,S,DV]
  float* W = out + (size_t)NB * NH * NS * NDV;         // [B,H,S,S]

  // Zero the 512 MB dense weight matrix once per call (capturable memset);
  // the kernel then writes only the non-zero band (~7.5% of it).
  hipMemsetAsync(W, 0, (size_t)NB * NH * NS * NS * sizeof(float), stream);

  dim3 grid(NS / 16, NH, NB);
  swa_sink_attn_kernel<<<grid, 32, 0, stream>>>(Q, K, V, O, W);
}
